// CrossAttention_71571335021222
// MI455X (gfx1250) — matrix operand order repaired and checked by reading, not compiled, not run on hardware
//
#include <hip/hip_runtime.h>
#include <hip/hip_bf16.h>

typedef __attribute__((ext_vector_type(8)))  __bf16 v8bf;
typedef __attribute__((ext_vector_type(16))) __bf16 v16bf;
typedef __attribute__((ext_vector_type(8)))  float  v8f;

// ---------------------------------------------------------------------------
// WMMA helpers (gfx1250 wave32: D[16x16,f32] = A[16x32,bf16] x B[32x16,bf16] + C)
// ---------------------------------------------------------------------------
__device__ __forceinline__ v8f wmma_bf16(v16bf a, v16bf b, v8f c) {
  return __builtin_amdgcn_wmma_f32_16x16x32_bf16(
      /*neg_a=*/false, a, /*neg_b=*/false, b,
      /*c_mod=*/(short)0, c, /*reuse_a=*/false, /*reuse_b=*/false);
}

// A-matrix fragment: lane L holds row m = L%16; with g = L/16 the 16 halves map
// to K = {g*8 .. g*8+7} then {16+g*8 .. 16+g*8+7}  (ISA 7.12.2, 16-bit A 16x32).
// rowPtr points at LDS row start (k-contiguous); rows are 16B aligned.
__device__ __forceinline__ v16bf load_frag_A(const __bf16* rowPtr, int g) {
  const v8bf lo = *(const v8bf*)(rowPtr + g * 8);
  const v8bf hi = *(const v8bf*)(rowPtr + 16 + g * 8);
  return __builtin_shufflevector(lo, hi, 0, 1, 2, 3, 4, 5, 6, 7,
                                 8, 9, 10, 11, 12, 13, 14, 15);
}

// B-matrix fragment: lane L holds col n = L%16; halves map to K = {g*8 .. g*8+7}
// then {16+g*8 .. 16+g*8+7} (same as A).
// rowPtr points at LDS row for this n (k-contiguous).
__device__ __forceinline__ v16bf load_frag_B(const __bf16* rowPtr, int g) {
  const v8bf lo = *(const v8bf*)(rowPtr + g * 8);
  const v8bf hi = *(const v8bf*)(rowPtr + 16 + g * 8);
  return __builtin_shufflevector(lo, hi, 0, 1, 2, 3, 4, 5, 6, 7,
                                 8, 9, 10, 11, 12, 13, 14, 15);
}

// ---------------------------------------------------------------------------
// GEMM: Y[b][o][n] = sum_c W[o][c] * X[b][c][n] + bias[o] (+ resid[b][o][n])
// Block tile 128(out) x 128(n), K step 32. 128 threads = 4 waves (2x2 of 64x64).
// ---------------------------------------------------------------------------
__global__ __launch_bounds__(128) void gemm_bf16_kernel(
    const float* __restrict__ W, const float* __restrict__ X,
    const float* __restrict__ bias, const float* __restrict__ resid,
    float* __restrict__ Y, int Cout, int Cin, int N,
    long xbs, long ybs, long rbs) {
  __shared__ __attribute__((aligned(16))) __bf16 As[128 * 40];  // [o][k], k contiguous
  __shared__ __attribute__((aligned(16))) __bf16 Bs[128 * 40];  // [n][k], k contiguous

  const int t = threadIdx.x;
  const int o0 = blockIdx.x * 128;
  const int n0 = blockIdx.y * 128;
  const int b  = blockIdx.z;
  const float* Xb = X + (long)b * xbs;
  float* Yb = Y + (long)b * ybs;

  const int w = t >> 5, lane = t & 31;
  const int l16 = lane & 15, g = lane >> 4;
  const int wr = (w >> 1) * 64;  // wave row offset in block tile
  const int wc = (w & 1) * 64;   // wave col offset

  v8f acc[4][4];
  const v8f vz = {0.f, 0.f, 0.f, 0.f, 0.f, 0.f, 0.f, 0.f};
#pragma unroll
  for (int mi = 0; mi < 4; ++mi)
#pragma unroll
    for (int ni = 0; ni < 4; ++ni) acc[mi][ni] = vz;

  for (int k0 = 0; k0 < Cin; k0 += 32) {
    __syncthreads();
    // A tile: W rows o0..o0+127, cols k0..k0+31 (coalesced reads).
#pragma unroll 4
    for (int e = 0; e < 32; ++e) {
      int i = e * 128 + t;
      int o = i >> 5, kk = i & 31;
      As[o * 40 + kk] = (__bf16)W[(long)(o0 + o) * Cin + (k0 + kk)];
    }
    // B tile transposed into [n][k]: thread t owns column n0+t (coalesced reads).
#pragma unroll 4
    for (int e = 0; e < 32; ++e) {
      Bs[t * 40 + e] = (__bf16)Xb[(long)(k0 + e) * N + (n0 + t)];
    }
    __syncthreads();

    v16bf aF[4], bF[4];
#pragma unroll
    for (int mi = 0; mi < 4; ++mi)
      aF[mi] = load_frag_A(&As[(wr + mi * 16 + l16) * 40], g);
#pragma unroll
    for (int ni = 0; ni < 4; ++ni)
      bF[ni] = load_frag_B(&Bs[(wc + ni * 16 + l16) * 40], g);
#pragma unroll
    for (int mi = 0; mi < 4; ++mi)
#pragma unroll
      for (int ni = 0; ni < 4; ++ni)
        acc[mi][ni] = wmma_bf16(aF[mi], bF[ni], acc[mi][ni]);
  }

  // Epilogue: C layout -> lane L: col n = l16, rows m = r + 8*g  (ISA 7.12.2)
#pragma unroll
  for (int mi = 0; mi < 4; ++mi) {
#pragma unroll
    for (int r = 0; r < 8; ++r) {
      int m = o0 + wr + mi * 16 + r + 8 * g;
      float bv = bias[m];
#pragma unroll
      for (int ni = 0; ni < 4; ++ni) {
        int n = n0 + wc + ni * 16 + l16;
        float val = acc[mi][ni][r] + bv;
        if (resid) val += resid[(long)b * rbs + (long)m * N + n];
        Yb[(long)m * N + n] = val;
      }
    }
  }
}

// ---------------------------------------------------------------------------
// Flash attention per (b,h): S = (Q^T K)/sqrt(D), online softmax, O = P V^T.
// Layout: Q/K/V/out are [B, C=512, N] with channel c = d*H + h (D=64, H=8).
// Grid: (N/64, B*H). 128 threads = 4 waves; each wave owns 16 queries.
// ---------------------------------------------------------------------------
__global__ __launch_bounds__(128) void attn_flash_kernel(
    const float* __restrict__ Q, const float* __restrict__ K,
    const float* __restrict__ V, float* __restrict__ O, int N) {
  const int H = 8;
  const int n0 = blockIdx.x * 64;
  const int bh = blockIdx.y;
  const int b = bh >> 3, h = bh & 7;
  const long base = (long)b * 512 * N;
  const float* q = Q + base;
  const float* k = K + base;
  const float* v = V + base;
  float* out = O + base;

  __shared__ __attribute__((aligned(16))) __bf16 Qs[64 * 72];      // [nq][d]
  __shared__ __attribute__((aligned(16))) __bf16 Ks[64 * 72];      // [m][d]
  __shared__ __attribute__((aligned(16))) __bf16 Vs[64 * 72];      // [d][m]
  __shared__ __attribute__((aligned(16))) __bf16 Ps[4][16 * 72];   // per-wave [nq][m]

  const int t = threadIdx.x;
  const int w = t >> 5, lane = t & 31;
  const int l16 = lane & 15, g = lane >> 4;

  // Stage Q tile transposed: Qs[nq][d]
#pragma unroll 4
  for (int e = 0; e < 32; ++e) {
    int i = e * 128 + t;
    int d = i >> 6, nq = i & 63;
    Qs[nq * 72 + d] = (__bf16)q[(long)(d * H + h) * N + (n0 + nq)];
  }

  float mrun[8], lrun[8];
  v8f o_acc[4];
  const v8f vz = {0.f, 0.f, 0.f, 0.f, 0.f, 0.f, 0.f, 0.f};
#pragma unroll
  for (int r = 0; r < 8; ++r) { mrun[r] = -3.0e38f; lrun[r] = 0.f; }
#pragma unroll
  for (int di = 0; di < 4; ++di) o_acc[di] = vz;

  for (int m0 = 0; m0 < N; m0 += 64) {
    __syncthreads();  // protect Ks/Vs (and Qs on first pass)
#pragma unroll 4
    for (int e = 0; e < 32; ++e) {
      int i = e * 128 + t;
      int d = i >> 6, ml = i & 63;
      long src = (long)(d * H + h) * N + (m0 + ml);
      Ks[ml * 72 + d] = (__bf16)k[src];  // K^T tile for B operand of S
      Vs[d * 72 + ml] = (__bf16)v[src];  // V tile (m contiguous) for B of O
    }
    __syncthreads();

    // S tile: 16 queries x 64 keys per wave
    v8f s[4];
#pragma unroll
    for (int ni = 0; ni < 4; ++ni) s[ni] = vz;
#pragma unroll
    for (int kd = 0; kd < 64; kd += 32) {
      v16bf aQ = load_frag_A(&Qs[(w * 16 + l16) * 72 + kd], g);
#pragma unroll
      for (int ni = 0; ni < 4; ++ni) {
        v16bf bK = load_frag_B(&Ks[(ni * 16 + l16) * 72 + kd], g);
        s[ni] = wmma_bf16(aQ, bK, s[ni]);
      }
    }
#pragma unroll
    for (int ni = 0; ni < 4; ++ni) s[ni] = s[ni] * 0.125f;  // 1/sqrt(64)

    // Online softmax per row (row nq = r + 8g spans the 16 lanes of group g)
#pragma unroll
    for (int r = 0; r < 8; ++r) {
      float mx = s[0][r];
#pragma unroll
      for (int ni = 1; ni < 4; ++ni) mx = fmaxf(mx, s[ni][r]);
#pragma unroll
      for (int off = 1; off < 16; off <<= 1)
        mx = fmaxf(mx, __shfl_xor(mx, off, 16));
      float mnew = fmaxf(mrun[r], mx);
      float corr = __expf(mrun[r] - mnew);
      float pv[4], rsum = 0.f;
#pragma unroll
      for (int ni = 0; ni < 4; ++ni) {
        pv[ni] = __expf(s[ni][r] - mnew);
        rsum += pv[ni];
      }
#pragma unroll
      for (int off = 1; off < 16; off <<= 1)
        rsum += __shfl_xor(rsum, off, 16);
      lrun[r] = lrun[r] * corr + rsum;
      mrun[r] = mnew;
#pragma unroll
      for (int di = 0; di < 4; ++di) o_acc[di][r] = o_acc[di][r] * corr;
      // Re-layout P via wave-private LDS (C-layout -> A-layout)
#pragma unroll
      for (int ni = 0; ni < 4; ++ni)
        Ps[w][(r + 8 * g) * 72 + ni * 16 + l16] = (__bf16)pv[ni];
    }

    // O += P * V^T  (A = P[16 x 64], B rows indexed by d, K = m)
#pragma unroll
    for (int kd = 0; kd < 64; kd += 32) {
      v16bf aP = load_frag_A(&Ps[w][l16 * 72 + kd], g);
#pragma unroll
      for (int di = 0; di < 4; ++di) {
        v16bf bV = load_frag_B(&Vs[(di * 16 + l16) * 72 + kd], g);
        o_acc[di] = wmma_bf16(aP, bV, o_acc[di]);
      }
    }
  }

  // Write out[b][c=d*H+h][n]; each lane writes 8 consecutive n per d-column.
#pragma unroll
  for (int di = 0; di < 4; ++di) {
    int c = (di * 16 + l16) * H + h;
#pragma unroll
    for (int r = 0; r < 8; ++r) {
      int n = n0 + w * 16 + r + 8 * g;
      out[(long)c * N + n] = o_acc[di][r] / lrun[r];
    }
  }
}

// ---------------------------------------------------------------------------
// Instance-norm over length dim (+ ReLU), in place. One block per (b, channel).
// ---------------------------------------------------------------------------
__global__ __launch_bounds__(256) void inorm_relu_kernel(float* __restrict__ x, int N) {
  float* p = x + (long)blockIdx.x * N;
  __shared__ float ssum[256], ssq[256];
  const int tid = threadIdx.x;
  float s = 0.f, qq = 0.f;
  for (int i = tid; i < N; i += 256) {
    float v = p[i];
    s += v;
    qq += v * v;
  }
  ssum[tid] = s;
  ssq[tid] = qq;
  __syncthreads();
  for (int off = 128; off > 0; off >>= 1) {
    if (tid < off) {
      ssum[tid] += ssum[tid + off];
      ssq[tid] += ssq[tid + off];
    }
    __syncthreads();
  }
  float mean = ssum[0] / (float)N;
  float var = ssq[0] / (float)N - mean * mean;
  float rs = rsqrtf(var + 1e-5f);
  for (int i = tid; i < N; i += 256) {
    float v = (p[i] - mean) * rs;
    p[i] = v > 0.f ? v : 0.f;
  }
}

// Copy src [B,512,N] into lower channel half of cat [B,1024,N].
__global__ void copy_lower_kernel(const float* __restrict__ src,
                                  float* __restrict__ dst, long CN, long C2N,
                                  long total) {
  long i = (long)blockIdx.x * blockDim.x + threadIdx.x;
  if (i < total) {
    long b = i / CN, rem = i % CN;
    dst[b * C2N + rem] = src[i];
  }
}

// ---------------------------------------------------------------------------
// Orchestration
// ---------------------------------------------------------------------------
extern "C" void kernel_launch(void* const* d_in, const int* in_sizes, int n_in,
                              void* d_out, int out_size, void* d_ws,
                              size_t ws_size, hipStream_t stream) {
  (void)in_sizes; (void)n_in; (void)out_size; (void)ws_size;
  const int B = 4, C = 512, N = 2048, H = 8;
  (void)H;
  const long CN = (long)C * N;       // per-batch stride of a [C,N] tensor
  const long BCN = (long)B * CN;

  const float* src = (const float*)d_in[0];
  const float* tgt = (const float*)d_in[1];
  const float* Wq = (const float*)d_in[2];  const float* bq = (const float*)d_in[3];
  const float* Wk = (const float*)d_in[4];  const float* bk = (const float*)d_in[5];
  const float* Wv = (const float*)d_in[6];  const float* bv = (const float*)d_in[7];
  const float* Wm = (const float*)d_in[8];  const float* bm = (const float*)d_in[9];
  const float* W1 = (const float*)d_in[10]; const float* b1 = (const float*)d_in[11];
  const float* W2 = (const float*)d_in[12]; const float* b2 = (const float*)d_in[13];

  float* ws = (float*)d_ws;
  // Workspace reuse (4*BCN floats ~ 67 MB):
  //   [0, 2*BCN)   : Q | K  during attention, then re-used as cat buffer
  //   [2*BCN,3*BCN): V      -+ re-used together as MLP hidden h1 (2*BCN)
  //   [3*BCN,4*BCN): att    -+
  float* catb = ws;
  float* Qb = ws;
  float* Kb = ws + BCN;
  float* Vb = ws + 2 * BCN;
  float* attb = ws + 3 * BCN;
  float* h1 = ws + 2 * BCN;

  float* outLo = (float*)d_out;
  float* outHi = outLo + BCN;

  auto half = [&](const float* x, const float* y, float* out) {
    dim3 blk(128);
    dim3 gC(C / 128, N / 128, B);
    dim3 g2C(2 * C / 128, N / 128, B);
    // Q from x; K,V from y
    gemm_bf16_kernel<<<gC, blk, 0, stream>>>(Wq, x, bq, nullptr, Qb, C, C, N, CN, CN, 0);
    gemm_bf16_kernel<<<gC, blk, 0, stream>>>(Wk, y, bk, nullptr, Kb, C, C, N, CN, CN, 0);
    gemm_bf16_kernel<<<gC, blk, 0, stream>>>(Wv, y, bv, nullptr, Vb, C, C, N, CN, CN, 0);
    attn_flash_kernel<<<dim3(N / 64, B * 8), blk, 0, stream>>>(Qb, Kb, Vb, attb, N);
    // cat = [x ; Wm*att + bm]  (Wm GEMM writes straight into upper half)
    copy_lower_kernel<<<(int)((BCN + 255) / 256), 256, 0, stream>>>(x, catb, CN, 2 * CN, BCN);
    gemm_bf16_kernel<<<gC, blk, 0, stream>>>(Wm, attb, bm, nullptr, catb + CN, C, C, N,
                                             CN, 2 * CN, 0);
    // MLP
    gemm_bf16_kernel<<<g2C, blk, 0, stream>>>(W1, catb, b1, nullptr, h1, 2 * C, 2 * C, N,
                                              2 * CN, 2 * CN, 0);
    inorm_relu_kernel<<<B * 2 * C, 256, 0, stream>>>(h1, N);
    // out = x + W2*h1 + b2
    gemm_bf16_kernel<<<gC, blk, 0, stream>>>(W2, h1, b2, x, out, C, 2 * C, N,
                                             2 * CN, CN, CN);
  };

  half(src, tgt, outLo);    // updated source
  half(tgt, outLo, outHi);  // updated target uses updated source
}
